// HyperbolicHead_61125974556677
// MI455X (gfx1250) — matrix operand order repaired and checked by reading, not compiled, not run on hardware
//
#include <hip/hip_runtime.h>
#include <hip/hip_bf16.h>
#include <cstdint>

typedef __attribute__((ext_vector_type(16))) _Float16 v16h;
typedef __attribute__((ext_vector_type(8)))  _Float16 v8h;
typedef __attribute__((ext_vector_type(8)))  float    v8f;

#define EPS_F      1e-6f
#define T_CLIP_F   0.96f
#define BALL_EPS_F 4e-3f
#define ATANH_E_F  1e-7f
#define ATANH_096  1.9459101090932196f   // atanh(0.96) = 0.5*ln(49)

__device__ __forceinline__ float softplus_f(float x) {
    return (x > 20.0f) ? x : log1pf(__expf(x));
}

__device__ __forceinline__ v16h cat8(v8h lo, v8h hi) {
    return __builtin_shufflevector(lo, hi, 0,1,2,3,4,5,6,7,8,9,10,11,12,13,14,15);
}

// ---- gfx1250 async global->LDS copy (ASYNCcnt path), with fallback --------
#define HAS_ASYNC_LDS __has_builtin(__builtin_amdgcn_global_load_async_to_lds_b128)

// Builtin signature (from hipcc diagnostic): param 0 is
//   'int __attribute__((vector_size(4*sizeof(int)))) __device__ *'  (addrspace(1) int4*)
typedef int v4i_gcc __attribute__((vector_size(4 * sizeof(int))));
typedef __attribute__((address_space(1))) v4i_gcc gbl_v4i;
typedef __attribute__((address_space(3))) v4i_gcc lds_v4i;

__device__ __forceinline__ void async_cp16(void* lds_dst, const void* gsrc) {
#if HAS_ASYNC_LDS
    __builtin_amdgcn_global_load_async_to_lds_b128(
        (gbl_v4i*)(uintptr_t)gsrc,
        (lds_v4i*)(uint32_t)(uintptr_t)lds_dst,
        0, 0);
#else
    (void)lds_dst; (void)gsrc;
#endif
}

__device__ __forceinline__ void wait_async0() {
#if __has_builtin(__builtin_amdgcn_s_wait_asynccnt)
    __builtin_amdgcn_s_wait_asynccnt(0);
#else
    asm volatile("s_wait_asynccnt 0x0" ::: "memory");
#endif
}

// ---------------------------------------------------------------------------
// prep_h: one block per cls row. Everything up to the GEMM collapses into a
// single scalar multiplier g(row) applied to cls, derived from ||cls_row||.
// Writes h (f16) and x2[row] = ||h_final||^2.
// ---------------------------------------------------------------------------
__global__ __launch_bounds__(256) void prep_h_kernel(
    const float* __restrict__ cls,
    const float* __restrict__ cls_scale_p,
    const float* __restrict__ curv_rho_p,
    _Float16* __restrict__ h,
    float* __restrict__ x2,
    int D)
{
    __shared__ float red[256];
    __shared__ float gmul;
    const int row = blockIdx.x;
    const int tid = threadIdx.x;
    const float* r = cls + (size_t)row * D;

    float v0 = r[tid];
    float v1 = r[tid + 256];
    float v2 = r[tid + 512];
    red[tid] = v0 * v0 + v1 * v1 + v2 * v2;
    __syncthreads();
    for (int s = 128; s > 0; s >>= 1) {
        if (tid < s) red[tid] += red[tid + s];
        __syncthreads();
    }

    if (tid == 0) {
        const float c   = softplus_f(*curv_rho_p);
        const float sc  = sqrtf(c);
        const float s_  = softplus_f(*cls_scale_p) + EPS_F;
        const float xn  = s_ * sqrtf(red[0]);          // ||x||
        const float xmx = ATANH_096 / (sc + EPS_F);    // pre-clip bound
        const float f1  = fminf(xmx / (xn + EPS_F), 1.0f);
        const float vn  = fmaxf(xn * f1, 1e-15f);      // clamped tangent norm
        const float tf  = tanhf(sc * vn) / (sc * vn);  // expmap0 factor
        const float hn  = tf * xn * f1;                // ||h|| after expmap0
        const float mxr = (1.0f - BALL_EPS_F) / sc;    // projx radius
        const float m2  = (hn > mxr) ? (mxr / hn) : 1.0f;
        const float hn2 = hn * m2;
        const float rmx = T_CLIP_F / (sc + EPS_F);     // post-clip radius
        const float f2  = fminf(rmx / (hn2 + EPS_F), 1.0f);
        gmul = s_ * f1 * tf * m2 * f2;
        const float hf = hn2 * f2;
        x2[row] = hf * hf;
    }
    __syncthreads();

    const float g = gmul;
    _Float16* hr = h + (size_t)row * D;
    hr[tid]       = (_Float16)(v0 * g);
    hr[tid + 256] = (_Float16)(v1 * g);
    hr[tid + 512] = (_Float16)(v2 * g);
}

// ---------------------------------------------------------------------------
// prep_p: projx of prototypes, padded to CP=1024 rows (rows >= C are zeroed
// so the GEMM needs no N-dimension guards). Writes p (f16) and y2[col].
// ---------------------------------------------------------------------------
__global__ __launch_bounds__(256) void prep_p_kernel(
    const float* __restrict__ proto,
    const float* __restrict__ curv_rho_p,
    _Float16* __restrict__ p,
    float* __restrict__ y2,
    int C, int D)
{
    __shared__ float red[256];
    __shared__ float gmul;
    const int row = blockIdx.x;
    const int tid = threadIdx.x;
    const bool live = (row < C);
    const float* r = proto + (size_t)row * D;

    float v0 = 0.f, v1 = 0.f, v2 = 0.f;
    if (live) { v0 = r[tid]; v1 = r[tid + 256]; v2 = r[tid + 512]; }
    red[tid] = v0 * v0 + v1 * v1 + v2 * v2;
    __syncthreads();
    for (int s = 128; s > 0; s >>= 1) {
        if (tid < s) red[tid] += red[tid + s];
        __syncthreads();
    }

    if (tid == 0) {
        if (live) {
            const float c   = softplus_f(*curv_rho_p);
            const float sc  = sqrtf(c);
            const float mxr = (1.0f - BALL_EPS_F) / sc;
            const float ss  = red[0];
            const float pn  = fmaxf(sqrtf(ss), 1e-15f);
            const float m   = (pn > mxr) ? (mxr / pn) : 1.0f;
            gmul = m;
            y2[row] = ss * m * m;
        } else {
            gmul = 0.0f;
            y2[row] = 0.0f;
        }
    }
    __syncthreads();

    const float g = gmul;
    _Float16* pr = p + (size_t)row * D;
    pr[tid]       = (_Float16)(v0 * g);
    pr[tid + 256] = (_Float16)(v1 * g);
    pr[tid + 512] = (_Float16)(v2 * g);
}

// ---------------------------------------------------------------------------
// WMMA GEMM (h @ p^T) + hyperbolic-distance epilogue.
// Block tile: 128(M) x 128(N), K step 32. 256 threads = 8 wave32 in a 2x4
// grid; each wave owns a 64x32 region = 4x2 v_wmma_f32_16x16x32_f16 tiles.
// Global->LDS staging uses async-to-LDS copies (ASYNCcnt) with LDS double
// buffering: copies for tile k+1 overlap the WMMAs of tile k.
// ---------------------------------------------------------------------------
#define BM 128
#define BN 128
#define BK 32
#define LDA 40   // half elements per LDS row (80B, 16B aligned)

__global__ __launch_bounds__(256) void gemm_hyp_kernel(
    const _Float16* __restrict__ h,
    const _Float16* __restrict__ p,
    const float* __restrict__ x2,
    const float* __restrict__ y2,
    const float* __restrict__ bias,
    const float* __restrict__ sigma_p,
    const float* __restrict__ curv_rho_p,
    float* __restrict__ out,
    int C, int D)
{
#if HAS_ASYNC_LDS
    __shared__ _Float16 As[2][BM][LDA];
    __shared__ _Float16 Bs[2][BN][LDA];
#else
    __shared__ _Float16 As[1][BM][LDA];
    __shared__ _Float16 Bs[1][BN][LDA];
#endif

    const int tid  = threadIdx.x;
    const int lane = tid & 31;
    const int wave = tid >> 5;
    const int wm   = wave >> 2;      // 0..1  (64 rows each)
    const int wn   = wave & 3;       // 0..3  (32 cols each)
    const int m0   = blockIdx.y * BM;
    const int n0   = blockIdx.x * BN;
    const int l    = lane & 15;
    const int half = lane >> 4;

    const v8f vzero = {0.f,0.f,0.f,0.f,0.f,0.f,0.f,0.f};
    v8f acc[4][2];
    #pragma unroll
    for (int t = 0; t < 4; ++t)
        #pragma unroll
        for (int u = 0; u < 2; ++u)
            acc[t][u] = vzero;

    // per-thread global->LDS mapping: 512 16B chunks per tile, 2 per thread
    const int rr0 = tid >> 2;          // row of chunk `tid`   (0..63)
    const int kk0 = (tid & 3) * 8;     // half-offset within row {0,8,16,24}
    const int nIter = D / BK;

#if HAS_ASYNC_LDS
    // ---- async double-buffered pipeline -----------------------------------
    {   // prologue: buffer 0 <- K window 0
        #pragma unroll
        for (int i = 0; i < 2; ++i) {
            const int rr = rr0 + 64 * i;
            async_cp16(&As[0][rr][kk0], h + (size_t)(m0 + rr) * D + kk0);
            async_cp16(&Bs[0][rr][kk0], p + (size_t)(n0 + rr) * D + kk0);
        }
    }
    for (int it = 0; it < nIter; ++it) {
        const int buf = it & 1;
        wait_async0();          // own copies into buf complete
        __syncthreads();        // everyone's copies done; buf^1 reads finished

        // A fragments: lanes 0-15 -> K 0..15; lanes 16-31 -> K 16..31
        const int ka = half ? 16 : 0;
        v16h afrag[4];
        #pragma unroll
        for (int t = 0; t < 4; ++t) {
            const int rr = wm * 64 + t * 16 + l;
            v8h lo = *(const v8h*)&As[buf][rr][ka];
            v8h hi = *(const v8h*)&As[buf][rr][ka + 8];
            afrag[t] = cat8(lo, hi);
        }
        // B fragments: lanes 0-15 -> K 0..15; lanes 16-31 -> K 16..31
        const int kb = half ? 16 : 0;
        v16h bfrag[2];
        #pragma unroll
        for (int u = 0; u < 2; ++u) {
            const int cc = wn * 32 + u * 16 + l;
            v8h lo = *(const v8h*)&Bs[buf][cc][kb];
            v8h hi = *(const v8h*)&Bs[buf][cc][kb + 8];
            bfrag[u] = cat8(lo, hi);
        }

        // kick off next tile into the other buffer (overlaps the WMMAs)
        if (it + 1 < nIter) {
            const int k1 = (it + 1) * BK;
            #pragma unroll
            for (int i = 0; i < 2; ++i) {
                const int rr = rr0 + 64 * i;
                async_cp16(&As[buf ^ 1][rr][kk0], h + (size_t)(m0 + rr) * D + k1 + kk0);
                async_cp16(&Bs[buf ^ 1][rr][kk0], p + (size_t)(n0 + rr) * D + k1 + kk0);
            }
        }

        #pragma unroll
        for (int t = 0; t < 4; ++t)
            #pragma unroll
            for (int u = 0; u < 2; ++u)
                acc[t][u] = __builtin_amdgcn_wmma_f32_16x16x32_f16(
                    false, afrag[t], false, bfrag[u],
                    (short)0, acc[t][u], false, false);
    }
#else
    // ---- fallback: single-buffered VGPR staging ---------------------------
    for (int k0 = 0; k0 < D; k0 += BK) {
        #pragma unroll
        for (int i = 0; i < 2; ++i) {
            const int rr = rr0 + 64 * i;
            *(v8h*)&As[0][rr][kk0] = *(const v8h*)(h + (size_t)(m0 + rr) * D + k0 + kk0);
            *(v8h*)&Bs[0][rr][kk0] = *(const v8h*)(p + (size_t)(n0 + rr) * D + k0 + kk0);
        }
        if (k0 + BK < D) {
            __builtin_prefetch(h + (size_t)(m0 + rr0) * D + k0 + BK + kk0, 0, 1);
            __builtin_prefetch(p + (size_t)(n0 + rr0) * D + k0 + BK + kk0, 0, 1);
        }
        __syncthreads();

        const int ka = half ? 16 : 0;
        v16h afrag[4];
        #pragma unroll
        for (int t = 0; t < 4; ++t) {
            const int rr = wm * 64 + t * 16 + l;
            afrag[t] = cat8(*(const v8h*)&As[0][rr][ka],
                            *(const v8h*)&As[0][rr][ka + 8]);
        }
        const int kb = half ? 16 : 0;
        v16h bfrag[2];
        #pragma unroll
        for (int u = 0; u < 2; ++u) {
            const int cc = wn * 32 + u * 16 + l;
            bfrag[u] = cat8(*(const v8h*)&Bs[0][cc][kb],
                            *(const v8h*)&Bs[0][cc][kb + 8]);
        }

        #pragma unroll
        for (int t = 0; t < 4; ++t)
            #pragma unroll
            for (int u = 0; u < 2; ++u)
                acc[t][u] = __builtin_amdgcn_wmma_f32_16x16x32_f16(
                    false, afrag[t], false, bfrag[u],
                    (short)0, acc[t][u], false, false);

        __syncthreads();
    }
#endif

    // ---------------- epilogue: hyperbolic distance -> logits ---------------
    const float c    = softplus_f(*curv_rho_p);
    const float sc   = sqrtf(c);
    const float sig  = fminf(fmaxf(softplus_f(*sigma_p) + EPS_F, 0.01f), 10.0f);
    const float i2s2 = 1.0f / (2.0f * sig * sig);
    const float dsc  = 2.0f / sc;
    const float zcl  = 1.0f - ATANH_E_F;

    #pragma unroll
    for (int u = 0; u < 2; ++u) {
        const int col = n0 + wn * 32 + u * 16 + l;
        if (col >= C) continue;
        const float yy = y2[col];
        const float bb = bias[col];
        #pragma unroll
        for (int t = 0; t < 4; ++t) {
            const int rbase = m0 + wm * 64 + t * 16 + half * 8;
            #pragma unroll
            for (int r = 0; r < 8; ++r) {
                const int rowi = rbase + r;
                const float xy = acc[t][u][r];
                const float xx = x2[rowi];
                const float A  = 1.0f - 2.0f * c * xy + c * yy;
                const float B  = 1.0f - c * xx;
                const float n2 = fmaxf(A * A * xx + B * B * yy - 2.0f * A * B * xy, 0.0f);
                const float dn = 1.0f - 2.0f * c * xy + c * c * xx * yy;
                const float mn = __fsqrt_rn(n2) / dn;
                const float z  = fminf(fmaxf(sc * mn, -zcl), zcl);
                const float d  = dsc * 0.5f * __logf((1.0f + z) / (1.0f - z));
                out[(size_t)rowi * C + col] = -(d * d) * i2s2 + bb;
            }
        }
    }
}

// ---------------------------------------------------------------------------
extern "C" void kernel_launch(void* const* d_in, const int* in_sizes, int n_in,
                              void* d_out, int out_size, void* d_ws, size_t ws_size,
                              hipStream_t stream) {
    const float* cls       = (const float*)d_in[0];
    const float* proto     = (const float*)d_in[1];
    const float* bias      = (const float*)d_in[2];
    const float* sigma     = (const float*)d_in[3];
    const float* cls_scale = (const float*)d_in[4];
    const float* curv_rho  = (const float*)d_in[5];

    const int D  = 768;
    const int N  = in_sizes[0] / D;   // 8192
    const int C  = in_sizes[1] / D;   // 1000
    const int CP = 1024;              // padded prototype count (BN multiple)

    char* ws = (char*)d_ws;
    size_t off = 0;
    _Float16* h = (_Float16*)(ws + off);  off += (size_t)N * D * sizeof(_Float16);
    off = (off + 255) & ~(size_t)255;
    _Float16* p = (_Float16*)(ws + off);  off += (size_t)CP * D * sizeof(_Float16);
    off = (off + 255) & ~(size_t)255;
    float* x2 = (float*)(ws + off);       off += (size_t)N * sizeof(float);
    off = (off + 255) & ~(size_t)255;
    float* y2 = (float*)(ws + off);

    prep_h_kernel<<<N, 256, 0, stream>>>(cls, cls_scale, curv_rho, h, x2, D);
    prep_p_kernel<<<CP, 256, 0, stream>>>(proto, curv_rho, p, y2, C, D);

    dim3 grid(CP / BN, N / BM);   // (8, 64)
    gemm_hyp_kernel<<<grid, 256, 0, stream>>>(h, p, x2, y2, bias, sigma, curv_rho,
                                              (float*)d_out, C, D);
}
